// BertAttentionDistance_89747636617542
// MI455X (gfx1250) — matrix operand order repaired and checked by reading, not compiled, not run on hardware
//
#include <hip/hip_runtime.h>
#include <hip/hip_bf16.h>

typedef __attribute__((ext_vector_type(16))) _Float16 v16h;
typedef __attribute__((ext_vector_type(8)))  _Float16 v8h;
typedef __attribute__((ext_vector_type(8)))  float    v8f;

#define H_DIM 1024
#define S_DIM 2048
#define B_DIM 4

__device__ inline v8f wmma_f16(v16h a, v16h b, v8f c) {
    return __builtin_amdgcn_wmma_f32_16x16x32_f16(false, a, false, b, (short)0, c, false, false);
}

// A operand: 16(M) x 32(K) f16, rows row0..row0+15 of row-major src (ld halves).
// Lane m / m+16 hold row m; halves 0..7 = K c0+bo..+7, halves 8..15 = K c0+16+bo..+23,
// bo = 8*(lane>=16). Two contiguous 16B loads.
__device__ inline v16h load_frag_a_f16(const _Float16* base, int row0, int ld, int c0, int lane) {
    int r  = row0 + (lane & 15);
    int bo = (lane >> 4) << 3;
    const _Float16* p = base + (size_t)r * ld + c0 + bo;
    v8h lo = *(const v8h*)p;
    v8h hi = *(const v8h*)(p + 16);
    return __builtin_shufflevector(lo, hi, 0,1,2,3,4,5,6,7,8,9,10,11,12,13,14,15);
}

// B operand: 32(K) x 16(N) f16 "column-major": lane n / n+16 hold column n (= row n of
// the transposed-operand matrix), halves 0..7 = K c0+bo..+7, halves 8..15 = K c0+16+bo..+23,
// bo = 8*(lane>=16). Two contiguous 16B loads.
// Using rows of Kmat/Vt/W gives the A@B^T form with row-major loads only.
__device__ inline v16h load_frag_b_f16(const _Float16* base, int row0, int ld, int c0, int lane) {
    int r  = row0 + (lane & 15);
    int ko = (lane >> 4) << 3;
    const _Float16* p = base + (size_t)r * ld + c0 + ko;
    v8h lo = *(const v8h*)p;
    v8h hi = *(const v8h*)(p + 16);
    return __builtin_shufflevector(lo, hi, 0,1,2,3,4,5,6,7,8,9,10,11,12,13,14,15);
}

// ---------------- Kernel 0: bulk fp32 -> f16 conversion ----------------
__global__ __launch_bounds__(256) void cvt_f16_kernel(
    const float* __restrict__ src, _Float16* __restrict__ dst, int n)
{
    int i = (blockIdx.x * 256 + threadIdx.x) * 8;
    if (i < n) {
#pragma unroll
        for (int j = 0; j < 8; ++j) dst[i + j] = (_Float16)src[i + j];
    }
}

// ---------------- Kernel 1: QKV projections (all-f16 operands) ----------------
// grid (512, 3), block 256. y: 0=Q, 1=K, 2=V(transposed store).
__global__ __launch_bounds__(256) void qkv_kernel(
    const _Float16* __restrict__ Xf,
    const _Float16* __restrict__ Wqf, const float* __restrict__ bq,
    const _Float16* __restrict__ Wkf, const float* __restrict__ bk,
    const _Float16* __restrict__ Wvf, const float* __restrict__ bv,
    _Float16* __restrict__ Qf, _Float16* __restrict__ Kf, _Float16* __restrict__ Vt)
{
    int lane = threadIdx.x & 31, wave = threadIdx.x >> 5;
    int n0 = blockIdx.x * 16;          // output row tile
    int j0 = wave * 128;               // this wave's 128 output cols
    int which = blockIdx.y;
    const _Float16* W  = (which == 0) ? Wqf : (which == 1) ? Wkf : Wvf;
    const float* bias  = (which == 0) ? bq  : (which == 1) ? bk  : bv;

    v8f acc[8];
#pragma unroll
    for (int s = 0; s < 8; ++s) acc[s] = (v8f){0,0,0,0,0,0,0,0};

    for (int c0 = 0; c0 < H_DIM; c0 += 32) {
        // issue all loads for this K-chunk before any WMMA: staged waits overlap
        v16h a = load_frag_a_f16(Xf, n0, H_DIM, c0, lane);
        v16h bfr[8];
#pragma unroll
        for (int s = 0; s < 8; ++s)
            bfr[s] = load_frag_b_f16(W, j0 + s * 16, H_DIM, c0, lane);
#pragma unroll
        for (int s = 0; s < 8; ++s)
            acc[s] = wmma_f16(a, bfr[s], acc[s]);
    }

    int hi = lane >> 4, nn = lane & 15;
#pragma unroll
    for (int s = 0; s < 8; ++s) {
        int j = j0 + s * 16 + nn;
        float bj = bias[j];
#pragma unroll
        for (int v = 0; v < 8; ++v) {
            int row = n0 + v + 8 * hi;
            float val = acc[s][v] + bj;
            if (which == 0) {
                Qf[(size_t)row * H_DIM + j] = (_Float16)val;
            } else if (which == 1) {
                Kf[(size_t)row * H_DIM + j] = (_Float16)val;
            } else {
                int b   = row >> 11;       // / S_DIM
                int tok = row & (S_DIM - 1);
                Vt[((size_t)b * H_DIM + j) * S_DIM + tok] = (_Float16)val;
            }
        }
    }
}

// ---------------- Kernel 2: wdd1[n] = dot(Q[n], dist_emb[1]) ----------------
// (mask_out==(rel==1) means only dist_emb row 1 ever contributes)
__global__ __launch_bounds__(256) void wdd1_kernel(
    const _Float16* __restrict__ Qf, const float* __restrict__ dist_emb,
    float* __restrict__ wdd1)
{
    __shared__ float red[8][32];
    int lane = threadIdx.x & 31, wave = threadIdx.x >> 5;
    int n = blockIdx.x * 8 + wave;
    const _Float16* q = Qf + (size_t)n * H_DIM;
    const float* de = dist_emb + H_DIM;   // row t = 1
    float s = 0.f;
    for (int d = lane; d < H_DIM; d += 32) s += (float)q[d] * de[d];
    red[wave][lane] = s;
    __syncthreads();
    if (lane == 0) {
        float t = 0.f;
        for (int i = 0; i < 32; ++i) t += red[wave][i];
        wdd1[n] = t;
    }
}

// ---------------- Kernel 3: flash attention with distance term ----------------
__global__ __launch_bounds__(256) void attn_kernel(
    const _Float16* __restrict__ Qf, const _Float16* __restrict__ Kf,
    const _Float16* __restrict__ Vt, const int* __restrict__ rel32,
    const float* __restrict__ amask, const float* __restrict__ wdd1,
    _Float16* __restrict__ Ctx)
{
    __shared__ float s_lds[16][128];
    __shared__ __align__(32) _Float16 p_lds[16][128];
    __shared__ float red[16][16];
    __shared__ float m_s[16], l_s[16], scl_s[16], wq_s[16];

    int tid = threadIdx.x, lane = tid & 31, wave = tid >> 5;
    int hi = lane >> 4, nn = lane & 15;
    int b   = blockIdx.x >> 7;
    int q0g = (blockIdx.x & 127) * 16;
    int n0  = b * S_DIM + q0g;

    if (tid < 16) { m_s[tid] = -1e30f; l_s[tid] = 0.f; wq_s[tid] = wdd1[n0 + tid]; }

    v8f acc[8];
#pragma unroll
    for (int s = 0; s < 8; ++s) acc[s] = (v8f){0,0,0,0,0,0,0,0};

    const float inv_sqrt = 1.0f / 32.0f;  // 1/sqrt(1024)
    const _Float16* Vtb = Vt + (size_t)b * H_DIM * S_DIM;
    const int* relbase  = rel32 + (size_t)n0 * S_DIM * 2;  // int64 low dwords
    __syncthreads();

    for (int k0 = 0; k0 < S_DIM; k0 += 128) {
        int kb = k0 + wave * 16;
        // prefetch next iteration's K rows / V columns for this wave
        if (k0 + 128 < S_DIM) {
            __builtin_prefetch(Kf + (size_t)(b * S_DIM + kb + 128) * H_DIM, 0, 1);
            __builtin_prefetch(Vtb + (size_t)(wave * 128) * S_DIM + (k0 + 128), 0, 1);
        }
        // --- scores: this wave's 16 keys, full H reduction (2x unrolled) ---
        v8f sc = (v8f){0,0,0,0,0,0,0,0};
        for (int c0 = 0; c0 < H_DIM; c0 += 64) {
            v16h a0 = load_frag_a_f16(Qf, n0, H_DIM, c0, lane);
            v16h b0 = load_frag_b_f16(Kf, b * S_DIM + kb, H_DIM, c0, lane);
            v16h a1 = load_frag_a_f16(Qf, n0, H_DIM, c0 + 32, lane);
            v16h b1 = load_frag_b_f16(Kf, b * S_DIM + kb, H_DIM, c0 + 32, lane);
            sc = wmma_f16(a0, b0, sc);
            sc = wmma_f16(a1, b1, sc);
        }
        int key  = kb + nn;
        float am = amask[b * S_DIM + key];
#pragma unroll
        for (int v = 0; v < 8; ++v) {
            int row = v + 8 * hi;
            int r = relbase[(size_t)row * S_DIM * 2 + (size_t)key * 2];
            float d = (r == 1) ? wq_s[row] : 0.f;
            s_lds[row][wave * 16 + nn] = (sc[v] + d) * inv_sqrt + am;
        }
        __syncthreads();

        // --- online softmax: partial max ---
        {
            int row = tid >> 4, g = tid & 15;
            float mx = s_lds[row][g * 8];
#pragma unroll
            for (int i = 1; i < 8; ++i) mx = fmaxf(mx, s_lds[row][g * 8 + i]);
            red[row][g] = mx;
        }
        __syncthreads();
        if (tid < 16) {
            float mx = red[tid][0];
#pragma unroll
            for (int i = 1; i < 16; ++i) mx = fmaxf(mx, red[tid][i]);
            float mo = m_s[tid];
            float mn = fmaxf(mo, mx);
            float sc2 = __expf(mo - mn);
            m_s[tid] = mn; scl_s[tid] = sc2; l_s[tid] *= sc2;
        }
        __syncthreads();

        // --- rescale running accumulators ---
#pragma unroll
        for (int s = 0; s < 8; ++s)
#pragma unroll
            for (int v = 0; v < 8; ++v) acc[s][v] *= scl_s[v + 8 * hi];

        // --- exp, f16 probs, partial row sums ---
        {
            int row = tid >> 4, g = tid & 15;
            float mn = m_s[row];
            float ps = 0.f;
#pragma unroll
            for (int i = 0; i < 8; ++i) {
                float p = __expf(s_lds[row][g * 8 + i] - mn);
                p_lds[row][g * 8 + i] = (_Float16)p;
                ps += p;
            }
            red[row][g] = ps;
        }
        __syncthreads();
        if (tid < 16) {
            float t = 0.f;
#pragma unroll
            for (int i = 0; i < 16; ++i) t += red[tid][i];
            l_s[tid] += t;
        }

        // --- PV: acc += P @ V (B operand = Vt rows, A@B^T form) ---
#pragma unroll
        for (int kk = 0; kk < 4; ++kk) {
            v16h a = load_frag_a_f16(&p_lds[0][0], 0, 128, kk * 32, lane);
            v16h bfr[8];
#pragma unroll
            for (int s = 0; s < 8; ++s)
                bfr[s] = load_frag_b_f16(Vtb, wave * 128 + s * 16, S_DIM, k0 + kk * 32, lane);
#pragma unroll
            for (int s = 0; s < 8; ++s)
                acc[s] = wmma_f16(a, bfr[s], acc[s]);
        }
    }
    __syncthreads();

    // --- finalize: divide by l, store ctx f16 ---
#pragma unroll
    for (int s = 0; s < 8; ++s) {
        int d = wave * 128 + s * 16 + nn;
#pragma unroll
        for (int v = 0; v < 8; ++v) {
            int row = v + 8 * hi;
            float o = acc[s][v] / l_s[row];
            Ctx[(size_t)(n0 + row) * H_DIM + d] = (_Float16)o;
        }
    }
}

// ---------------- Kernel 4: output projection + residual + LayerNorm ----------------
__global__ __launch_bounds__(256) void out_kernel(
    const _Float16* __restrict__ Ctx, const _Float16* __restrict__ Wof,
    const float* __restrict__ bo, const float* __restrict__ hidden,
    const float* __restrict__ ln_g, const float* __restrict__ ln_b,
    float* __restrict__ out)
{
    __shared__ float x_lds[16][H_DIM];
    __shared__ float red[16][16], red2[16][16];
    __shared__ float mu_s[16], rs_s[16];
    int tid = threadIdx.x, lane = tid & 31, wave = tid >> 5;
    int n0 = blockIdx.x * 16;
    int j0 = wave * 128;

    v8f acc[8];
#pragma unroll
    for (int s = 0; s < 8; ++s) acc[s] = (v8f){0,0,0,0,0,0,0,0};

    for (int c0 = 0; c0 < H_DIM; c0 += 32) {
        v16h a = load_frag_a_f16(Ctx, n0, H_DIM, c0, lane);
        v16h bfr[8];
#pragma unroll
        for (int s = 0; s < 8; ++s)
            bfr[s] = load_frag_b_f16(Wof, j0 + s * 16, H_DIM, c0, lane);
#pragma unroll
        for (int s = 0; s < 8; ++s)
            acc[s] = wmma_f16(a, bfr[s], acc[s]);
    }

    int hi = lane >> 4, nn = lane & 15;
#pragma unroll
    for (int s = 0; s < 8; ++s) {
        int j = j0 + s * 16 + nn;
        float bj = bo[j];
#pragma unroll
        for (int v = 0; v < 8; ++v) {
            int row = v + 8 * hi;
            x_lds[row][j] = acc[s][v] + bj + hidden[(size_t)(n0 + row) * H_DIM + j];
        }
    }
    __syncthreads();
    {
        int row = tid >> 4, g = tid & 15;
        float s1 = 0.f, s2 = 0.f;
        for (int i = 0; i < 64; ++i) {
            float x = x_lds[row][g * 64 + i];
            s1 += x; s2 += x * x;
        }
        red[row][g] = s1; red2[row][g] = s2;
    }
    __syncthreads();
    if (tid < 16) {
        float s1 = 0.f, s2 = 0.f;
#pragma unroll
        for (int i = 0; i < 16; ++i) { s1 += red[tid][i]; s2 += red2[tid][i]; }
        float mu  = s1 / (float)H_DIM;
        float var = s2 / (float)H_DIM - mu * mu;
        mu_s[tid] = mu;
        rs_s[tid] = rsqrtf(var + 1e-12f);
    }
    __syncthreads();
    {
        int row = tid >> 4, g = tid & 15;
        float mu = mu_s[row], rs = rs_s[row];
        for (int i = 0; i < 64; ++i) {
            int j = g * 64 + i;
            out[(size_t)(n0 + row) * H_DIM + j] = (x_lds[row][j] - mu) * rs * ln_g[j] + ln_b[j];
        }
    }
}

extern "C" void kernel_launch(void* const* d_in, const int* in_sizes, int n_in,
                              void* d_out, int out_size, void* d_ws, size_t ws_size,
                              hipStream_t stream) {
    (void)in_sizes; (void)n_in; (void)out_size; (void)ws_size;
    const float* hidden = (const float*)d_in[0];
    const float* amask  = (const float*)d_in[1];
    const int*   rel32  = (const int*)d_in[2];   // int64 viewed as dword pairs (values < 32)
    const float* Wq = (const float*)d_in[3];  const float* bq = (const float*)d_in[4];
    const float* Wk = (const float*)d_in[5];  const float* bk = (const float*)d_in[6];
    const float* Wv = (const float*)d_in[7];  const float* bv = (const float*)d_in[8];
    const float* dist = (const float*)d_in[9];
    const float* Wo = (const float*)d_in[10]; const float* bo = (const float*)d_in[11];
    const float* lng = (const float*)d_in[12]; const float* lnb = (const float*)d_in[13];

    char* ws = (char*)d_ws;
    const size_t MB = 1024 * 1024;
    _Float16* Xf   = (_Float16*)(ws + 0 * MB);
    _Float16* Qf   = (_Float16*)(ws + 16 * MB);
    _Float16* Kf   = (_Float16*)(ws + 32 * MB);
    _Float16* Vt   = (_Float16*)(ws + 48 * MB);
    _Float16* Ctx  = (_Float16*)(ws + 64 * MB);
    _Float16* Wqf  = (_Float16*)(ws + 80 * MB);
    _Float16* Wkf  = (_Float16*)(ws + 82 * MB);
    _Float16* Wvf  = (_Float16*)(ws + 84 * MB);
    _Float16* Wof  = (_Float16*)(ws + 86 * MB);
    float*    wdd1 = (float*)(ws + 88 * MB);

    const int NX = B_DIM * S_DIM * H_DIM;   // 8388608
    const int NW = H_DIM * H_DIM;           // 1048576
    cvt_f16_kernel<<<NX / (256 * 8), 256, 0, stream>>>(hidden, Xf, NX);
    cvt_f16_kernel<<<NW / (256 * 8), 256, 0, stream>>>(Wq, Wqf, NW);
    cvt_f16_kernel<<<NW / (256 * 8), 256, 0, stream>>>(Wk, Wkf, NW);
    cvt_f16_kernel<<<NW / (256 * 8), 256, 0, stream>>>(Wv, Wvf, NW);
    cvt_f16_kernel<<<NW / (256 * 8), 256, 0, stream>>>(Wo, Wof, NW);

    qkv_kernel<<<dim3(512, 3), 256, 0, stream>>>(Xf, Wqf, bq, Wkf, bk, Wvf, bv, Qf, Kf, Vt);
    wdd1_kernel<<<1024, 256, 0, stream>>>(Qf, dist, wdd1);
    attn_kernel<<<512, 256, 0, stream>>>(Qf, Kf, Vt, rel32, amask, wdd1, Ctx);
    out_kernel<<<512, 256, 0, stream>>>(Ctx, Wof, bo, hidden, lng, lnb, (float*)d_out);
}
